// EuclideanAttention_88476326298147
// MI455X (gfx1250) — matrix operand order repaired and checked by reading, not compiled, not run on hardware
//
#include <hip/hip_runtime.h>

// ---------------------------------------------------------------------------
// EuclideanAttention on gfx1250 (MI455X):
//   phase 1: bf16 convert x + fused [Wq*scale; Wk; Wv; Wqv; pad] weights
//   phase 2: WMMA GEMM  QKV[8192,1600] = xbf @ Bqkv^T        (v_wmma bf16)
//   phase 3: wave-per-(node,head) edge softmax / aggregation  (VALU+shfl)
//   phase 4: WMMA GEMM  out[8192,512] = featbf @ Wout^T + b   (v_wmma bf16)
// Wave tile 32x64 (2 A-frags x 4 B-frags -> 8 WMMA / K-step), block 256x64.
// ---------------------------------------------------------------------------

#define N_    8192
#define D_    512
#define H_    8
#define DK_   64
#define DEG_  16
#define E_    (N_ * DEG_)
#define QKV_LD 1600               // 512 q | 512 k | 512 v | 24 qv | 40 pad
#define FEAT_  544                // H * (DK + 4)

typedef __attribute__((ext_vector_type(16))) __bf16       v16bf;
typedef __attribute__((ext_vector_type(8)))  float        v8f;
typedef __attribute__((ext_vector_type(4)))  unsigned int v4u;

union FragU { v16bf bf; v4u u[2]; };

__device__ __forceinline__ unsigned short f2bf(float f) {
  unsigned int u = __float_as_uint(f);
  u += 0x7FFFu + ((u >> 16) & 1u);          // round-to-nearest-even
  return (unsigned short)(u >> 16);
}

// A-matrix fragment, 16-bit 16x32 (ISA 7.12.2):
//   lane<16 : row=lane,    K = {0..7} then {16..23}
//   lane>=16: row=lane-16, K = {8..15} then {24..31}
__device__ __forceinline__ v16bf load_frag_a(const unsigned short* base, int ld, int lane) {
  const int r = lane & 15;
  const int k = (lane >> 4) << 3;           // 0 or 8
  const unsigned short* p = base + (size_t)r * ld + k;
  FragU f;
  f.u[0] = *(const v4u*)(p);
  f.u[1] = *(const v4u*)(p + 16);
  return f.bf;
}

// B-matrix fragment, 16-bit 32x16 (rows striped across lanes):
//   lane<16 : col=lane,    K = {0..7} then {16..23}
//   lane>=16: col=lane-16, K = {8..15} then {24..31}
// B memory is W row-major [Ncol, K], so B[k, n] = W[n, k] -> lane n reads W row n.
__device__ __forceinline__ v16bf load_frag_b(const unsigned short* base, int ld, int lane) {
  const int r = lane & 15;
  const int k = (lane >> 4) << 3;           // 0 or 8
  const unsigned short* p = base + (size_t)r * ld + k;
  FragU f;
  f.u[0] = *(const v4u*)(p);
  f.u[1] = *(const v4u*)(p + 16);
  return f.bf;
}

// C = A[M,K] * B[Ncol,K]^T (+bias), fp32 out. Block 256 thr = 8 waves stacked
// along M; wave tile 32x64 -> block tile 256x64. M%256==0, Ncol%64==0, K%32==0.
__global__ __launch_bounds__(256) void gemm_bf16_nt(
    const unsigned short* __restrict__ A, const unsigned short* __restrict__ B,
    float* __restrict__ C, const float* __restrict__ bias,
    int M, int Ncol, int K) {
  const int lane = threadIdx.x & 31;
  const int wave = threadIdx.x >> 5;
  const int rowBase = blockIdx.y * 256 + wave * 32;
  const int colBase = blockIdx.x * 64;
  const unsigned short* Ab = A + (size_t)rowBase * K;
  const unsigned short* Bb = B + (size_t)colBase * K;

  v8f acc[2][4] = {};
  for (int k = 0; k < K; k += 32) {
    __builtin_prefetch(Ab + k + 64, 0, 3);
    __builtin_prefetch(Bb + k + 64, 0, 3);
    v16bf a0 = load_frag_a(Ab + k, K, lane);
    v16bf a1 = load_frag_a(Ab + (size_t)16 * K + k, K, lane);
    v16bf b0 = load_frag_b(Bb + k, K, lane);
    v16bf b1 = load_frag_b(Bb + (size_t)16 * K + k, K, lane);
    v16bf b2 = load_frag_b(Bb + (size_t)32 * K + k, K, lane);
    v16bf b3 = load_frag_b(Bb + (size_t)48 * K + k, K, lane);
    acc[0][0] = __builtin_amdgcn_wmma_f32_16x16x32_bf16(false, a0, false, b0, (short)0, acc[0][0], false, false);
    acc[0][1] = __builtin_amdgcn_wmma_f32_16x16x32_bf16(false, a0, false, b1, (short)0, acc[0][1], false, false);
    acc[0][2] = __builtin_amdgcn_wmma_f32_16x16x32_bf16(false, a0, false, b2, (short)0, acc[0][2], false, false);
    acc[0][3] = __builtin_amdgcn_wmma_f32_16x16x32_bf16(false, a0, false, b3, (short)0, acc[0][3], false, false);
    acc[1][0] = __builtin_amdgcn_wmma_f32_16x16x32_bf16(false, a1, false, b0, (short)0, acc[1][0], false, false);
    acc[1][1] = __builtin_amdgcn_wmma_f32_16x16x32_bf16(false, a1, false, b1, (short)0, acc[1][1], false, false);
    acc[1][2] = __builtin_amdgcn_wmma_f32_16x16x32_bf16(false, a1, false, b2, (short)0, acc[1][2], false, false);
    acc[1][3] = __builtin_amdgcn_wmma_f32_16x16x32_bf16(false, a1, false, b3, (short)0, acc[1][3], false, false);
  }

  // 16x16 f32 C/D layout: VGPR r -> M=r (lane<16) / M=r+8 (lane>=16), N=lane&15
  const int rOff = (lane >> 4) << 3;
  const int cOff = lane & 15;
#pragma unroll
  for (int mi = 0; mi < 2; ++mi) {
#pragma unroll
    for (int ni = 0; ni < 4; ++ni) {
      const int col = colBase + ni * 16 + cOff;
      const float bb = bias ? bias[col] : 0.0f;
#pragma unroll
      for (int r = 0; r < 8; ++r) {
        C[(size_t)(rowBase + mi * 16 + rOff + r) * Ncol + col] = acc[mi][ni][r] + bb;
      }
    }
  }
}

// One wave per (node, head). Lanes span DK via float2 (k = 2*lane, 2*lane+1).
__global__ __launch_bounds__(256) void edge_attn(
    const float* __restrict__ QKV,
    const long long* __restrict__ col_index,
    const long long* __restrict__ to_col,
    const float* __restrict__ att_bias,
    const float* __restrict__ dist,
    const float* __restrict__ pos,
    const float* __restrict__ col_pos,
    unsigned short* __restrict__ featbf) {
  const int lane = threadIdx.x & 31;
  const int wave = threadIdx.x >> 5;
  const int pair = blockIdx.x * 8 + wave;           // N_*H_ pairs total
  const int n = pair >> 3;                          // H_ == 8
  const int h = pair & 7;

  const float* qrow = QKV + (size_t)n * QKV_LD + h * DK_;
  const float2 q = *(const float2*)(qrow + 2 * lane);         // pre-scaled by 1/sqrt(dk)
  const float* qvp = QKV + (size_t)n * QKV_LD + 1536 + h * 3;
  const float qv0 = qvp[0], qv1 = qvp[1], qv2 = qvp[2];
  const float px = pos[n * 3 + 0], py = pos[n * 3 + 1], pz = pos[n * 3 + 2];

  float lg[DEG_], invd[DEG_];
  int   tn[DEG_], cc[DEG_];
#pragma unroll
  for (int j = 0; j < DEG_; ++j) {
    const int e = n * DEG_ + j;
    const int c = (int)col_index[e];
    const int t = (int)to_col[c];
    cc[j] = c; tn[j] = t;
    const float2 kv = *(const float2*)(QKV + (size_t)t * QKV_LD + 512 + h * DK_ + 2 * lane);
    float d = q.x * kv.x + q.y * kv.y;
#pragma unroll
    for (int s = 16; s >= 1; s >>= 1) d += __shfl_xor(d, s, 32);  // wave32 reduce
    const float rx = col_pos[c * 3 + 0] - px;
    const float ry = col_pos[c * 3 + 1] - py;
    const float rz = col_pos[c * 3 + 2] - pz;
    const float dd = dist[e];
    const float id = dd > 0.0f ? 1.0f / dd : 0.0f;  // dist==0 -> inf -> terms vanish
    invd[j] = id;
    lg[j] = d + att_bias[(size_t)h * E_ + e] + (qv0 * rx + qv1 * ry + qv2 * rz) * id;
  }

  float m = lg[0];
#pragma unroll
  for (int j = 1; j < DEG_; ++j) m = fmaxf(m, lg[j]);
  float s = 0.0f;
#pragma unroll
  for (int j = 0; j < DEG_; ++j) { lg[j] = __expf(lg[j] - m); s += lg[j]; }
  const float rs = 1.0f / s;

  float a0 = 0.0f, a1 = 0.0f, dvx = 0.0f, dvy = 0.0f, dvz = 0.0f, aid = 0.0f;
#pragma unroll
  for (int j = 0; j < DEG_; ++j) {
    const float w  = lg[j] * rs;
    const float nw = w * invd[j];
    const float2 vv = *(const float2*)(QKV + (size_t)tn[j] * QKV_LD + 1024 + h * DK_ + 2 * lane);
    a0 += w * vv.x;  a1 += w * vv.y;
    dvx += nw * col_pos[cc[j] * 3 + 0];
    dvy += nw * col_pos[cc[j] * 3 + 1];
    dvz += nw * col_pos[cc[j] * 3 + 2];
    aid += nw;
  }

  const float fx = dvx - aid * px, fy = dvy - aid * py, fz = dvz - aid * pz;
  const float nrm = sqrtf(fx * fx + fy * fy + fz * fz);
  const float in  = 1.0f / fmaxf(nrm, 1e-12f);

  unsigned short* frow = featbf + (size_t)n * FEAT_ + h * (DK_ + 4);
  frow[2 * lane]     = f2bf(a0);
  frow[2 * lane + 1] = f2bf(a1);
  if (lane == 0) {
    frow[64] = f2bf(fx * in);
    frow[65] = f2bf(fy * in);
    frow[66] = f2bf(fz * in);
    frow[67] = f2bf(aid);
  }
}

__global__ void cvt_bf16_scaled(const float* __restrict__ s, unsigned short* __restrict__ d,
                                int n, float scale) {
  for (int i = blockIdx.x * blockDim.x + threadIdx.x; i < n; i += gridDim.x * blockDim.x)
    d[i] = f2bf(s[i] * scale);
}

__global__ void fill_u16(unsigned short* __restrict__ d, int n, unsigned short v) {
  for (int i = blockIdx.x * blockDim.x + threadIdx.x; i < n; i += gridDim.x * blockDim.x)
    d[i] = v;
}

extern "C" void kernel_launch(void* const* d_in, const int* in_sizes, int n_in,
                              void* d_out, int out_size, void* d_ws, size_t ws_size,
                              hipStream_t stream) {
  const float*     x         = (const float*)d_in[0];
  // d_in[1] = row_index (unused: row_index[e] == e / DEG by construction)
  const long long* col_index = (const long long*)d_in[2];
  const long long* to_col    = (const long long*)d_in[3];
  const float*     att_bias  = (const float*)d_in[4];
  const float*     dist      = (const float*)d_in[5];
  const float*     pos       = (const float*)d_in[6];
  const float*     col_pos   = (const float*)d_in[7];
  const float*     Wq        = (const float*)d_in[8];
  const float*     Wqv       = (const float*)d_in[9];
  const float*     Wk        = (const float*)d_in[10];
  const float*     Wv        = (const float*)d_in[11];
  const float*     Wout      = (const float*)d_in[12];
  const float*     bout      = (const float*)d_in[13];
  float*           out       = (float*)d_out;

  char* p = (char*)d_ws;
  auto alloc = [&](size_t bytes) -> void* {
    void* r = (void*)p;
    p += (bytes + 255) & ~(size_t)255;
    return r;
  };
  unsigned short* xbf   = (unsigned short*)alloc((size_t)N_ * D_ * 2);
  unsigned short* Bqkv  = (unsigned short*)alloc((size_t)QKV_LD * D_ * 2);
  float*          QKV   = (float*)alloc((size_t)N_ * QKV_LD * 4);
  unsigned short* featb = (unsigned short*)alloc((size_t)N_ * FEAT_ * 2);
  unsigned short* Woutb = (unsigned short*)alloc((size_t)D_ * FEAT_ * 2);

  const float scale = 0.125f;  // 1/sqrt(DK_)

  // phase 1: bf16 conversions (scale folded into Wq)
  cvt_bf16_scaled<<<2048, 256, 0, stream>>>(x,   xbf,                N_ * D_,        1.0f);
  cvt_bf16_scaled<<<512,  256, 0, stream>>>(Wq,  Bqkv,               H_ * DK_ * D_,  scale);
  cvt_bf16_scaled<<<512,  256, 0, stream>>>(Wk,  Bqkv + 512 * D_,    H_ * DK_ * D_,  1.0f);
  cvt_bf16_scaled<<<512,  256, 0, stream>>>(Wv,  Bqkv + 1024 * D_,   H_ * DK_ * D_,  1.0f);
  cvt_bf16_scaled<<<48,   256, 0, stream>>>(Wqv, Bqkv + 1536 * D_,   H_ * 3 * D_,    1.0f);
  fill_u16<<<80, 256, 0, stream>>>(Bqkv + 1560 * D_, 40 * D_, (unsigned short)0);
  cvt_bf16_scaled<<<512,  256, 0, stream>>>(Wout, Woutb,             D_ * FEAT_,     1.0f);

  // phase 2: fused q|k|v|qv projection GEMM  (8192 x 1600 x 512)
  gemm_bf16_nt<<<dim3(QKV_LD / 64, N_ / 256), 256, 0, stream>>>(
      xbf, Bqkv, QKV, nullptr, N_, QKV_LD, D_);

  // phase 3: per-row segment softmax + aggregation (wave per node-head)
  edge_attn<<<N_ * H_ / 8, 256, 0, stream>>>(
      QKV, col_index, to_col, att_bias, dist, pos, col_pos, featb);

  // phase 4: output projection GEMM  (8192 x 512 x 544) + bias
  gemm_bf16_nt<<<dim3(D_ / 64, N_ / 256), 256, 0, stream>>>(
      featb, Woutb, out, bout, N_, D_, FEAT_);
}
